// Block_4440996184125
// MI455X (gfx1250) — matrix operand order repaired and checked by reading, not compiled, not run on hardware
//
#include <hip/hip_runtime.h>
#include <math.h>

// ---------------- types for WMMA ----------------
typedef __bf16 v16bf __attribute__((ext_vector_type(16)));
typedef float  v8f   __attribute__((ext_vector_type(8)));
typedef unsigned int uint4v __attribute__((ext_vector_type(4)));

// ---------------- problem constants ----------------
#define WDIM  8192
#define BN    8
#define CN    256
#define EN    512
#define GN    4
#define INNERC 1024
#define WT    32          // output positions per block
#define HP    8           // halo on each side (need >=2; 8 keeps 16-multiples)
#define XT    48          // positions of x' computed per block (WT + 2*HP)
#define NTILES 3          // XT / 16

// ---------------- ws (scratch) layout ----------------
// bf16 (ushort) section
#define WS_WQ   0           // 256*64
#define WS_WK   16384
#define WS_WV   32768
#define WS_WP   49152
#define WS_WR0  65536       // 1024*192 (k = tap*64 + in)
#define WS_WR1  262144      // 256*768  (k = tap*256 + in)
#define WS_U16_TOTAL 458752 // ushorts -> 917504 bytes
// float section (offsets in floats from float base)
#define WSF_CC   0          // 8*256
#define WSF_CC2  2048       // 8*1024
#define WSF_AXA  10240      // 8*4
#define WSF_AYA  10272
#define WSF_AXR  10304
#define WSF_AYR  10336

// ---------------- LDS layout (bytes) ----------------
#define SM_XS   0           // float  xs [XT][CN]        49152
#define SM_YB   49152       // ushort yb [XT][CN]        24576  (y, later x' bf16)
#define SM_Y2B  73728       // ushort y2b[XT][INNERC]    98304
#define SM_SCR  172032      // 65536 scratch union:
                            //   phase A: float xstage[CN][XT] (49152, async-load target)
                            //   phase B: qs[16][CN] f32 | ks | vs | yo[16][CN] bf16
                            //   phase C: y2t[16][INNERC] f32 (65536)
#define SM_CC   237568      // float cc[256]
#define SM_CC2  238592      // float cc2[1024]
#define SM_TL   242688      // float tl[16]  (axA,ayA,axR,ayR per group)
#define SM_RED  242752      // float red[256] (C2 partial sums)
#define SMEM_BYTES 243776

static __device__ __forceinline__ unsigned short f2bf(float f) {
  union { float f; unsigned int u; } v; v.f = f;
  unsigned int u = v.u;
  unsigned int r = u + 0x7fffu + ((u >> 16) & 1u);   // round-to-nearest-even
  return (unsigned short)(r >> 16);
}

union FragU { uint4v q[2]; v16bf v; };

// A-matrix 16x32 bf16 fragment (ISA 7.12.2): lanes 0-15: M=lane, K=0..7|16..23;
// lanes 16-31: M=lane-16, K=8..15|24..31.  row = &W[m][0], lo = (lane<16?0:8)
static __device__ __forceinline__ v16bf load_a_frag(const unsigned short* row, int k0, int lo) {
  FragU f;
  f.q[0] = *(const uint4v*)(row + k0 + lo);
  f.q[1] = *(const uint4v*)(row + k0 + 16 + lo);
  return f.v;
}
// B-matrix 32x16 bf16 fragment, same K order as A at fixed N
// (lanes 0-15: K=k0..k0+7|k0+16..k0+23, N=lane; lanes 16-31: K=k0+8..k0+15|k0+24..k0+31, N=lane-16)
static __device__ __forceinline__ v16bf load_b_frag(const unsigned short* src) {
  FragU f;
  f.q[0] = *(const uint4v*)(src);
  f.q[1] = *(const uint4v*)(src + 16);
  return f.v;
}
static __device__ __forceinline__ v8f wmma_bf16(v16bf a, v16bf b, v8f c) {
  return __builtin_amdgcn_wmma_f32_16x16x32_bf16(false, a, false, b, (short)0, c, false, false);
}

// =====================================================================
// Precompute 1: EDM2 weight normalization -> bf16, WMMA-A row-major.
// rows 0..1023: wq/wk/wv/wproj (K=64); 1024..2047: res0 (K=192, permute);
// 2048..2303: res1 (K=768, permute)
// =====================================================================
__global__ void norm_weights_kernel(const float* __restrict__ w_q,
                                    const float* __restrict__ w_k,
                                    const float* __restrict__ w_v,
                                    const float* __restrict__ w_proj,
                                    const float* __restrict__ w_res0,
                                    const float* __restrict__ w_res1,
                                    unsigned short* __restrict__ wsu) {
  int id = blockIdx.x * blockDim.x + threadIdx.x;
  if (id >= 2304) return;
  const float* src; unsigned short* dst; int K; int inner = 0;
  if (id < 1024) {
    int fam = id >> 8, row = id & 255; K = 64;
    const float* w = (fam == 0) ? w_q : (fam == 1) ? w_k : (fam == 2) ? w_v : w_proj;
    src = w + row * 64; dst = wsu + fam * 16384 + row * 64;
  } else if (id < 2048) {
    int row = id - 1024; K = 192; inner = 64;
    src = w_res0 + row * 192; dst = wsu + WS_WR0 + row * 192;
  } else {
    int row = id - 2048; K = 768; inner = 256;
    src = w_res1 + row * 768; dst = wsu + WS_WR1 + row * 768;
  }
  float ss = 0.f;
  for (int k = 0; k < K; ++k) { float v = src[k]; ss += v * v; }
  float sk = sqrtf((float)K);
  float sc = 1.f / (sk * (1e-4f + sqrtf(ss) / sk));   // normalize * 1/sqrt(fan_in), gain=1
  if (inner == 0) {
    for (int k = 0; k < K; ++k) dst[k] = f2bf(src[k] * sc);
  } else {
    for (int in = 0; in < inner; ++in)
      for (int tap = 0; tap < 3; ++tap)
        dst[tap * inner + in] = f2bf(src[in * 3 + tap] * sc);
  }
}

// =====================================================================
// Precompute 2: emb-conditioned scalars: c (per-channel), c2, balance coeffs.
// =====================================================================
__global__ void emb_cond_kernel(const float* __restrict__ emb,
                                const float* __restrict__ w_qkv_emb,
                                const float* __restrict__ gain_qkv,
                                const float* __restrict__ w_emb,
                                const float* __restrict__ emb_gain,
                                const float* __restrict__ w_bal_attn,
                                const float* __restrict__ g_bal_attn,
                                const float* __restrict__ w_bal_res,
                                const float* __restrict__ g_bal_res,
                                float* __restrict__ wsf) {
  int id = blockIdx.x * blockDim.x + threadIdx.x;
  if (id >= 10304) return;
  if (id < 2048) {                       // cc: grouped 1x1 conv of emb, +1
    int b = id >> 8, o = id & 255, g = o >> 6;
    const float* wr = w_qkv_emb + o * 128;
    const float* e  = emb + b * EN + g * 128;
    float ss = 0.f, dot = 0.f;
    for (int k = 0; k < 128; ++k) { float v = wr[k]; ss += v * v; }
    float sk = sqrtf(128.f);
    float sc = gain_qkv[0] / (sk * (1e-4f + sqrtf(ss) / sk));
    for (int k = 0; k < 128; ++k) dot += wr[k] * e[k];
    wsf[WSF_CC + id] = dot * sc + 1.f;
  } else if (id < 10240) {               // cc2
    int t = id - 2048;
    int b = t >> 10, o = t & 1023, g = o >> 8;
    const float* wr = w_emb + o * 128;
    const float* e  = emb + b * EN + g * 128;
    float ss = 0.f, dot = 0.f;
    for (int k = 0; k < 128; ++k) { float v = wr[k]; ss += v * v; }
    float sk = sqrtf(128.f);
    float sc = emb_gain[0] / (sk * (1e-4f + sqrtf(ss) / sk));
    for (int k = 0; k < 128; ++k) dot += wr[k] * e[k];
    wsf[WSF_CC2 + t] = dot * sc + 1.f;
  } else {                               // balance coefficients
    int t = id - 10240;
    int b = t >> 3, r = t & 7, which = r >> 2, g = r & 3;
    const float* wr = ((which == 0) ? w_bal_attn : w_bal_res) + g * EN;
    float gain = (which == 0) ? g_bal_attn[0] : g_bal_res[0];
    const float* e = emb + b * EN;
    float ss = 0.f, dot = 0.f;
    for (int k = 0; k < EN; ++k) { float v = wr[k]; ss += v * v; }
    float sk = sqrtf((float)EN);
    float sc = gain / (sk * (1e-4f + sqrtf(ss) / sk));
    for (int k = 0; k < EN; ++k) dot += wr[k] * e[k];
    float logit = dot * sc - 2.0f;       // + OFFSET
    float tt = 1.f / (1.f + __expf(-logit));
    tt = fminf(fmaxf(tt, 0.01f), 0.99f);
    float rs = rsqrtf((1.f - tt) * (1.f - tt) + tt * tt);
    int idx = b * 4 + g;
    if (which == 0) { wsf[WSF_AXA + idx] = (1.f - tt) * rs; wsf[WSF_AYA + idx] = tt * rs; }
    else            { wsf[WSF_AXR + idx] = (1.f - tt) * rs; wsf[WSF_AYR + idx] = tt * rs; }
  }
}

// =====================================================================
// Fused main kernel: one block = (batch b, 32-wide W tile + halo 8)
// =====================================================================
__global__ __launch_bounds__(256) void fused_mp_block(
    const float* __restrict__ x,
    const unsigned short* __restrict__ wqn,
    const unsigned short* __restrict__ wkn,
    const unsigned short* __restrict__ wvn,
    const unsigned short* __restrict__ wpn,
    const unsigned short* __restrict__ wr0n,
    const unsigned short* __restrict__ wr1n,
    const float* __restrict__ cc_all,
    const float* __restrict__ cc2_all,
    const float* __restrict__ axA,
    const float* __restrict__ ayA,
    const float* __restrict__ axR,
    const float* __restrict__ ayR,
    float* __restrict__ out) {
  const int tid  = threadIdx.x;
  const int lane = tid & 31;
  const int wid  = tid >> 5;
  const int b    = blockIdx.y;
  const int pos0 = (int)blockIdx.x * WT - HP;

  // per-lane WMMA fragment roles
  const int n    = lane & 15;                 // N (position within tile)
  const int lo   = (lane < 16) ? 0 : 8;       // A-frag K sub-offset
  const int koff = (lane < 16) ? 0 : 8;       // B-frag K offset
  const int mrow = (lane < 16) ? 0 : 8;       // C-frag M offset

  extern __shared__ char smem[];
  float*          xs  = (float*)(smem + SM_XS);
  unsigned short* yb  = (unsigned short*)(smem + SM_YB);
  unsigned short* y2b = (unsigned short*)(smem + SM_Y2B);
  char*           scr = smem + SM_SCR;
  float*          stg = (float*)(scr);        // phase A staging [CN][XT]
  float*          qs  = (float*)(scr);
  float*          ks  = (float*)(scr + 16384);
  float*          vs  = (float*)(scr + 32768);
  unsigned short* yo  = (unsigned short*)(scr + 49152);
  float*          y2t = (float*)(scr);        // overlaps (phase C)
  float*          cc  = (float*)(smem + SM_CC);
  float*          cc2 = (float*)(smem + SM_CC2);
  float*          tl  = (float*)(smem + SM_TL);
  float*          red = (float*)(smem + SM_RED);

  // ---- load conditioning scalars ----
  cc[tid] = cc_all[b * CN + tid];
  for (int i = tid; i < INNERC; i += 256) cc2[i] = cc2_all[b * INNERC + i];
  if (tid < 4) {
    tl[tid]      = axA[b * 4 + tid];
    tl[4 + tid]  = ayA[b * 4 + tid];
    tl[8 + tid]  = axR[b * 4 + tid];
    tl[12 + tid] = ayR[b * 4 + tid];
  }

  // ---- phase A1: stage x tile [ch][XT] into LDS ----
  const bool interior = (pos0 >= 0) && (pos0 + XT <= WDIM);
  if (interior) {
    // CDNA5 async global->LDS copy (GVS mode), tracked by ASYNCcnt.
    // Each channel row = 48 floats = 12 x 16B segments, lanes 0..11 active.
    for (int ch = wid; ch < CN; ch += 8) {
      if (lane < 12) {
        unsigned ldsoff = (unsigned)(SM_SCR + (ch * XT) * 4 + lane * 16);
        unsigned voff   = (unsigned)(lane * 16);
        unsigned long long base =
            (unsigned long long)(const void*)(x + ((size_t)b * CN + ch) * (size_t)WDIM + pos0);
        asm volatile("global_load_async_to_lds_b128 %0, %1, %2"
                     :: "v"(ldsoff), "v"(voff), "s"(base) : "memory");
      }
    }
    asm volatile("s_wait_asynccnt 0x0" ::: "memory");
  } else {
    for (int ch = wid; ch < CN; ch += 8) {
      const float* xrow = x + ((size_t)b * CN + ch) * (size_t)WDIM;
      for (int j = lane; j < XT; j += 32) {
        int gp = pos0 + j;
        stg[ch * XT + j] = (gp >= 0 && gp < WDIM) ? xrow[gp] : 0.f;
      }
    }
  }
  __syncthreads();

  // ---- phase A2: transpose to position-major, y = x*c (bf16) ----
  for (int ch = wid; ch < CN; ch += 8) {
    const float cch = cc[ch];
    for (int j = lane; j < XT; j += 32) {
      float xv = stg[ch * XT + j];
      xs[j * CN + ch] = xv;
      yb[j * CN + ch] = f2bf(xv * cch);
    }
  }
  __syncthreads();

  // ---- phase B: qkv -> normalize -> 4x4 attention -> proj -> balance ----
  for (int t = 0; t < NTILES; ++t) {
    // B1: q/k/v grouped 64x64 GEMMs (WMMA)
    for (int item = wid; item < 48; item += 8) {
      int proj = item >> 4, g = (item >> 2) & 3, mt = item & 3;
      const unsigned short* wb = (proj == 0) ? wqn : (proj == 1) ? wkn : wvn;
      const unsigned short* row = wb + (g * 64 + mt * 16 + n) * 64;
      v8f acc = {};
      for (int kk = 0; kk < 2; ++kk) {
        int k0 = kk * 32;
        v16bf a  = load_a_frag(row, k0, lo);
        v16bf bb = load_b_frag(yb + (t * 16 + n) * CN + g * 64 + k0 + koff);
        acc = wmma_bf16(a, bb, acc);
      }
      float* dst = ((proj == 0) ? qs : (proj == 1) ? ks : vs)
                   + n * CN + g * 64 + mt * 16 + mrow;
      *(v8f*)dst = acc;
    }
    __syncthreads();
    // B2: normalize q,k,v per (pos, group) over 64 channels (d=64)
    if (tid < 192) {
      int p = tid / 12, r = tid % 12, pr = r >> 2, g = r & 3;
      float* buf = ((pr == 0) ? qs : (pr == 1) ? ks : vs) + p * CN + g * 64;
      float ss = 0.f;
      for (int c2i = 0; c2i < 64; ++c2i) { float v = buf[c2i]; ss += v * v; }
      float sc = 1.f / (1e-4f + sqrtf(ss) * 0.125f);
      for (int c2i = 0; c2i < 64; ++c2i) buf[c2i] *= sc;
    }
    __syncthreads();
    // B3: per-position 4x4 attention (rows independent under softmax)
    if (tid < 64) {
      int p = tid >> 2, g = tid & 3;
      const float* qrow = qs + p * CN + g * 64;
      float sco[4];
      for (int h = 0; h < 4; ++h) {
        const float* krow = ks + p * CN + h * 64;
        float d = 0.f;
        for (int c2i = 0; c2i < 64; ++c2i) d += qrow[c2i] * krow[c2i];
        sco[h] = d * 0.125f;               // 1/sqrt(64)
      }
      float mx = fmaxf(fmaxf(sco[0], sco[1]), fmaxf(sco[2], sco[3]));
      float e[4], sum = 0.f;
      for (int h = 0; h < 4; ++h) { e[h] = __expf(sco[h] - mx); sum += e[h]; }
      float inv = 1.f / sum;
      for (int h = 0; h < 4; ++h) e[h] *= inv;
      for (int c2i = 0; c2i < 64; ++c2i) {
        float o = 0.f;
        for (int h = 0; h < 4; ++h) o += e[h] * vs[p * CN + h * 64 + c2i];
        yo[p * CN + g * 64 + c2i] = f2bf(o);
      }
    }
    __syncthreads();
    // B4: proj GEMM + adaptive balance -> x' (f32 in xs, bf16 in yb)
    for (int item = wid; item < 16; item += 8) {
      int g = item >> 2, mt = item & 3;
      const unsigned short* row = wpn + (g * 64 + mt * 16 + n) * 64;
      v8f acc = {};
      for (int kk = 0; kk < 2; ++kk) {
        int k0 = kk * 32;
        v16bf a  = load_a_frag(row, k0, lo);
        v16bf bb = load_b_frag(yo + n * CN + g * 64 + k0 + koff);
        acc = wmma_bf16(a, bb, acc);
      }
      int j = t * 16 + n;
      int chb = g * 64 + mt * 16 + mrow;
      float ax = tl[g], ay = tl[4 + g];
      float* xr = xs + j * CN + chb;
      unsigned short* ybr = yb + j * CN + chb;
#pragma unroll
      for (int v = 0; v < 8; ++v) {
        float nx = xr[v] * ax + acc[v] * ay;
        xr[v] = nx;
        ybr[v] = f2bf(nx);
      }
    }
    __syncthreads();
  }

  // ---- phase C: res0 (K=192 GEMM) * c2 -> group-norm -> SiLU -> bf16 ----
  for (int t = 0; t < NTILES; ++t) {
    for (int item = wid; item < 64; item += 8) {
      int g = item >> 4, mt = item & 15;
      const unsigned short* row = wr0n + (g * 256 + mt * 16 + n) * 192;
      v8f acc = {};
      for (int kk = 0; kk < 6; ++kk) {
        int k0 = kk * 32;
        int tap = k0 >> 6, kin = k0 & 63;
        int jj = t * 16 + n + tap - 1;
        jj = (jj < 0) ? 0 : ((jj >= XT) ? XT - 1 : jj);   // edge garbage unused
        v16bf a  = load_a_frag(row, k0, lo);
        v16bf bb = load_b_frag(yb + jj * CN + g * 64 + kin + koff);
        acc = wmma_bf16(a, bb, acc);
      }
      *(v8f*)(y2t + n * INNERC + g * 256 + mt * 16 + mrow) = acc;
    }
    __syncthreads();
    // C2: c2 scale -> group-normalize (d=256) -> SiLU/0.596 -> bf16
    // 4 threads per (pos, group): partial sums over 64-channel quarters.
    {
      int p = tid >> 4, g = (tid >> 2) & 3, q = tid & 3;
      int j = t * 16 + p, gp = pos0 + j;
      const float* src = y2t + p * INNERC + g * 256 + q * 64;
      const float* c2g = cc2 + g * 256 + q * 64;
      float ss = 0.f;
      for (int ic = 0; ic < 64; ++ic) { float v = src[ic] * c2g[ic]; ss += v * v; }
      red[tid] = ss;
      __syncthreads();
      int base = tid & ~3;
      float tot = red[base] + red[base + 1] + red[base + 2] + red[base + 3];
      float sc = 1.f / (1e-4f + sqrtf(tot) * 0.0625f);
      unsigned short* drow = y2b + j * INNERC + g * 256 + q * 64;
      if (gp < 0 || gp >= WDIM) {
        for (int ic = 0; ic < 64; ++ic) drow[ic] = 0;     // explicit conv zero-pad
      } else {
        for (int ic = 0; ic < 64; ++ic) {
          float z = src[ic] * c2g[ic] * sc;
          float s = z / (1.f + __expf(-z));
          drow[ic] = f2bf(s * 1.6778523f);                // 1/0.596
        }
      }
    }
    __syncthreads();
  }

  // ---- phase D: res1 (K=768 GEMM) + balance + clip -> global out ----
  for (int unit = wid; unit < 32; unit += 8) {
    int ot = unit >> 4, item = unit & 15;
    int g = item >> 2, mt = item & 3;
    int j0 = HP + ot * 16;
    const unsigned short* row = wr1n + (g * 64 + mt * 16 + n) * 768;
    v8f acc = {};
    for (int kk = 0; kk < 24; ++kk) {
      int k0 = kk * 32;
      int tap = k0 >> 8, kin = k0 & 255;
      int jj = j0 + n + tap - 1;                          // 7..40, in range
      v16bf a  = load_a_frag(row, k0, lo);
      v16bf bb = load_b_frag(y2b + jj * INNERC + g * 256 + kin + koff);
      acc = wmma_bf16(a, bb, acc);
    }
    int j = j0 + n;
    int gp = pos0 + j;                                    // always in [0, WDIM)
    int chb = g * 64 + mt * 16 + mrow;
    float ax = tl[8 + g], ay = tl[12 + g];
    const float* xr = xs + j * CN + chb;
    float* obase = out + (size_t)b * CN * WDIM + gp;
#pragma unroll
    for (int v = 0; v < 8; ++v) {
      float val = xr[v] * ax + acc[v] * ay;
      val = fminf(fmaxf(val, -256.f), 256.f);
      obase[(size_t)(chb + v) * WDIM] = val;
    }
  }
}

// =====================================================================
extern "C" void kernel_launch(void* const* d_in, const int* in_sizes, int n_in,
                              void* d_out, int out_size, void* d_ws, size_t ws_size,
                              hipStream_t stream) {
  (void)in_sizes; (void)n_in; (void)out_size; (void)ws_size;
  const float* x          = (const float*)d_in[0];
  const float* emb        = (const float*)d_in[1];
  const float* w_qkv_emb  = (const float*)d_in[2];
  const float* gain_qkv   = (const float*)d_in[3];
  const float* w_q        = (const float*)d_in[4];
  const float* w_k        = (const float*)d_in[5];
  const float* w_v        = (const float*)d_in[6];
  const float* w_proj     = (const float*)d_in[7];
  const float* w_res0     = (const float*)d_in[8];
  const float* w_emb      = (const float*)d_in[9];
  const float* emb_gain   = (const float*)d_in[10];
  const float* w_res1     = (const float*)d_in[11];
  const float* w_bal_attn = (const float*)d_in[12];
  const float* g_bal_attn = (const float*)d_in[13];
  const float* w_bal_res  = (const float*)d_in[14];
  const float* g_bal_res  = (const float*)d_in[15];

  unsigned short* wsu = (unsigned short*)d_ws;
  float* wsf = (float*)((char*)d_ws + (size_t)WS_U16_TOTAL * sizeof(unsigned short));

  norm_weights_kernel<<<9, 256, 0, stream>>>(w_q, w_k, w_v, w_proj, w_res0, w_res1, wsu);
  emb_cond_kernel<<<41, 256, 0, stream>>>(emb, w_qkv_emb, gain_qkv, w_emb, emb_gain,
                                          w_bal_attn, g_bal_attn, w_bal_res, g_bal_res, wsf);

  dim3 grid(WDIM / WT, BN);
  fused_mp_block<<<grid, 256, SMEM_BYTES, stream>>>(
      x,
      wsu + WS_WQ, wsu + WS_WK, wsu + WS_WV, wsu + WS_WP,
      wsu + WS_WR0, wsu + WS_WR1,
      wsf + WSF_CC, wsf + WSF_CC2,
      wsf + WSF_AXA, wsf + WSF_AYA, wsf + WSF_AXR, wsf + WSF_AYR,
      (float*)d_out);
}
